// NeighborhoodAttention2D_84911503442058
// MI455X (gfx1250) — matrix operand order repaired and checked by reading, not compiled, not run on hardware
//
#include <hip/hip_runtime.h>

typedef __bf16 bf16;
typedef __attribute__((ext_vector_type(16))) __bf16 v16bf;
typedef __attribute__((ext_vector_type(8)))  __bf16 v8bf;
typedef __attribute__((ext_vector_type(8)))  float  v8f;

static_assert(sizeof(bf16) == 2, "bf16 size");

__device__ __forceinline__ v8f wmma_bf16(v16bf a, v16bf b, v8f c) {
    // D = A(16x32 bf16) * B(32x16 bf16) + C(16x16 f32)
    return __builtin_amdgcn_wmma_f32_16x16x32_bf16(false, a, false, b,
                                                   (short)0, c, false, false);
}

__device__ __forceinline__ v16bf combine16(v8bf lo, v8bf hi) {
    v16bf a;
#pragma unroll
    for (int e = 0; e < 8; ++e) { a[e] = lo[e]; a[e + 8] = hi[e]; }
    return a;
}

// ---------------------------------------------------------------------------
// fp32 -> bf16 elementwise convert (weights)
// ---------------------------------------------------------------------------
__global__ __launch_bounds__(256) void f32_to_bf16_kernel(
    const float* __restrict__ in, bf16* __restrict__ out, int n) {
    int i = blockIdx.x * 256 + threadIdx.x;
    if (i < n) out[i] = (bf16)in[i];
}

// ---------------------------------------------------------------------------
// WMMA GEMM:  C[m,n] = sum_k A[m,k] * Bw[n,k] + bias[n]
// Block tile 128(M) x 128(N), 256 threads = 8 waves arranged 4(M) x 2(N).
// Each wave computes 32x64 = 8 WMMA accumulators; K stepped by 32 through
// double-buffered LDS with register-staged global loads (latency overlap)
// and global_prefetch at distance 2.
// A_IS_F32: stage-convert A from fp32 while loading into LDS.
// OUT_F32:  write fp32 (final output) vs bf16 (intermediate).
// ---------------------------------------------------------------------------
template <bool A_IS_F32, bool OUT_F32>
__global__ __launch_bounds__(256) void gemm_wmma_kernel(
    const void* __restrict__ Aptr, int lda,
    const bf16* __restrict__ Bw, int ldb,
    const float* __restrict__ bias,
    void* __restrict__ Out, int ldo, int K) {
    __shared__ __align__(32) bf16 As[2][128][32];   // 16 KB
    __shared__ __align__(32) bf16 Bs[2][128][32];   // 16 KB

    const int t    = threadIdx.x;
    const int w    = t >> 5;         // wave id 0..7 (wave32)
    const int lane = t & 31;
    const int half = lane >> 4;      // 0: lanes 0-15, 1: lanes 16-31
    const int nl   = lane & 15;
    const int wm   = w & 3;          // wave M group (4 x 32 rows)
    const int wn   = w >> 2;         // wave N group (2 x 64 cols)
    const long Mbase = (long)blockIdx.y * 128;
    const long Nbase = (long)blockIdx.x * 128;

    v8f acc[2][4];
#pragma unroll
    for (int mt = 0; mt < 2; ++mt)
#pragma unroll
        for (int nt = 0; nt < 4; ++nt)
#pragma unroll
            for (int r = 0; r < 8; ++r) acc[mt][nt][r] = 0.0f;

    // Register staging buffers: 512 chunks of 8 elements for A and for B,
    // 2 chunks per thread each. chunk cid -> row = cid>>2, k-part = cid&3.
    float4 fa[2][2];
    v8bf   ra[2];
    v8bf   rb[2];

    auto ldg = [&](int kt) {
        const long gk = (long)kt * 32;
#pragma unroll
        for (int q = 0; q < 2; ++q) {
            const int cid  = t + q * 256;
            const int row  = cid >> 2;
            const int part = cid & 3;
            if (A_IS_F32) {
                const float* ap =
                    (const float*)Aptr + (Mbase + row) * lda + gk + part * 8;
                fa[q][0] = *(const float4*)(ap);
                fa[q][1] = *(const float4*)(ap + 4);
            } else {
                ra[q] = *(const v8bf*)((const bf16*)Aptr +
                                       (Mbase + row) * lda + gk + part * 8);
            }
            rb[q] = *(const v8bf*)(Bw + (Nbase + row) * ldb + gk + part * 8);
        }
    };
    auto sts = [&](int buf) {
#pragma unroll
        for (int q = 0; q < 2; ++q) {
            const int cid  = t + q * 256;
            const int row  = cid >> 2;
            const int part = cid & 3;
            if (A_IS_F32) {
                v8bf v;
                v[0] = (bf16)fa[q][0].x; v[1] = (bf16)fa[q][0].y;
                v[2] = (bf16)fa[q][0].z; v[3] = (bf16)fa[q][0].w;
                v[4] = (bf16)fa[q][1].x; v[5] = (bf16)fa[q][1].y;
                v[6] = (bf16)fa[q][1].z; v[7] = (bf16)fa[q][1].w;
                *(v8bf*)&As[buf][row][part * 8] = v;
            } else {
                *(v8bf*)&As[buf][row][part * 8] = ra[q];
            }
            *(v8bf*)&Bs[buf][row][part * 8] = rb[q];
        }
    };

    const int nk = K >> 5;
    ldg(0);
    int buf = 0;
    for (int kt = 0; kt < nk; ++kt) {
        sts(buf);
        __syncthreads();
        if (kt + 1 < nk) ldg(kt + 1);
        if (kt + 2 < nk) {
            // keep HBM ahead of the WMMA pipe: global_prefetch_b8
            const int row = t >> 2, part = t & 3;
            const long gk = (long)(kt + 2) * 32 + part * 8;
            if (A_IS_F32)
                __builtin_prefetch(
                    (const float*)Aptr + (Mbase + row) * lda + gk, 0, 0);
            else
                __builtin_prefetch(
                    (const bf16*)Aptr + (Mbase + row) * lda + gk, 0, 0);
            __builtin_prefetch(Bw + (Nbase + row) * ldb + gk, 0, 0);
        }
        // ---- 8 WMMAs from the current buffer ----
#pragma unroll
        for (int mt = 0; mt < 2; ++mt) {
            v8bf lo = *(const v8bf*)&As[buf][wm * 32 + mt * 16 + nl][16 * half];
            v8bf hi = *(const v8bf*)&As[buf][wm * 32 + mt * 16 + nl][16 * half + 8];
            v16bf a = combine16(lo, hi);
#pragma unroll
            for (int nt = 0; nt < 4; ++nt) {
                v16bf b =
                    *(const v16bf*)&Bs[buf][wn * 64 + nt * 16 + nl][16 * half];
                acc[mt][nt] = wmma_bf16(a, b, acc[mt][nt]);
            }
        }
        buf ^= 1;
    }

    // ---- epilogue: C/D layout M = 8*half + r, N = lane%16 ----
#pragma unroll
    for (int mt = 0; mt < 2; ++mt) {
#pragma unroll
        for (int nt = 0; nt < 4; ++nt) {
#pragma unroll
            for (int r = 0; r < 8; ++r) {
                long gm = Mbase + wm * 32 + mt * 16 + 8 * half + r;
                long gn = Nbase + wn * 64 + nt * 16 + nl;
                float val = acc[mt][nt][r] + bias[gn];
                if (OUT_F32)
                    ((float*)Out)[gm * ldo + gn] = val;
                else
                    ((bf16*)Out)[gm * ldo + gn] = (bf16)val;
            }
        }
    }
}

// ---------------------------------------------------------------------------
// Neighborhood attention: one block per (batch, head, 4x4 pixel tile).
// 10x10 clamped halo; logits via WMMA, per-pixel 49-way softmax, AV via WMMA.
// 128 threads = 4 waves.
// ---------------------------------------------------------------------------
__global__ __launch_bounds__(128) void natten_wmma_kernel(
    const bf16* __restrict__ qkv,   // [32768, 2304] bf16: [q|k|v] x head x dim
    bf16* __restrict__ aout) {      // [32768, 768] bf16
    __shared__ __align__(32) bf16 Qs[16][64];    // pixel x dim
    __shared__ __align__(32) bf16 Ks[112][64];   // halo pos x dim (pad >=100 zero)
    __shared__ __align__(32) bf16 Vt[64][128];   // dim x halo pos (pad zero)
    __shared__ __align__(32) bf16 At[16][128];   // pixel x halo prob (zero outside win)
    __shared__ float Lg[16][112];                // pixel x halo logits

    const int t    = threadIdx.x;
    const int w    = t >> 5;
    const int lane = t & 31;
    const int half = lane >> 4;
    const int nl   = lane & 15;

    int bid = blockIdx.x;
    const int tjx  = bid & 15; bid >>= 4;
    const int tix  = bid & 15; bid >>= 4;
    const int head = bid % 12;
    const int b    = bid / 12;

    const int ti = tix * 4, tj = tjx * 4;
    const int hbi = min(max(ti - 3, 0), 54);   // halo base (10 wide, clamped)
    const int hbj = min(max(tj - 3, 0), 54);
    const long cofs = (long)head * 64;

    // ---- stage Q (16x64): 128 chunks of 8 bf16, one per thread ----
    {
        int p = t >> 3, c = t & 7;
        int i = ti + (p >> 2), j = tj + (p & 3);
        long row = ((long)(b * 64 + i) * 64 + j);
        *(uint4*)&Qs[p][c * 8] = *(const uint4*)(qkv + row * 2304 + cofs + c * 8);
    }
    // ---- stage K halo (100 rows, zero rows 100..111) ----
    for (int cid = t; cid < 112 * 8; cid += 128) {
        int pos = cid >> 3, c = cid & 7;
        if (pos < 100) {
            int i = hbi + pos / 10, j = hbj + pos % 10;
            long row = ((long)(b * 64 + i) * 64 + j);
            *(uint4*)&Ks[pos][c * 8] =
                *(const uint4*)(qkv + row * 2304 + 768 + cofs + c * 8);
        } else {
            uint4 z = {0, 0, 0, 0};
            *(uint4*)&Ks[pos][c * 8] = z;
        }
    }
    // ---- zero Vt and At ----
    for (int cid = t; cid < (64 * 128 + 16 * 128) / 8; cid += 128) {
        uint4 z = {0, 0, 0, 0};
        bf16* base = (cid < 64 * 128 / 8) ? &Vt[0][0] : &At[0][0];
        int off = (cid < 64 * 128 / 8) ? cid * 8 : (cid - 64 * 128 / 8) * 8;
        *(uint4*)(base + off) = z;
    }
    __syncthreads();
    // ---- stage V halo, transposed into Vt[dim][pos] ----
    for (int cid = t; cid < 100 * 8; cid += 128) {
        int pos = cid >> 3, c = cid & 7;
        int i = hbi + pos / 10, j = hbj + pos % 10;
        long row = ((long)(b * 64 + i) * 64 + j);
        v8bf v = *(const v8bf*)(qkv + row * 2304 + 1536 + cofs + c * 8);
#pragma unroll
        for (int e = 0; e < 8; ++e) Vt[c * 8 + e][pos] = v[e];
    }
    __syncthreads();

    // ---- logits = scale * Q(16x64) @ K^T(64x112); 7 N-tiles over 4 waves ----
    v16bf aq[2];
#pragma unroll
    for (int s = 0; s < 2; ++s) {
        v8bf lo = *(const v8bf*)&Qs[nl][s * 32 + 16 * half];
        v8bf hi = *(const v8bf*)&Qs[nl][s * 32 + 16 * half + 8];
        aq[s] = combine16(lo, hi);
    }
    for (int nt = w; nt < 7; nt += 4) {     // wave-uniform trip count
        v8f acc = {0, 0, 0, 0, 0, 0, 0, 0};
#pragma unroll
        for (int s = 0; s < 2; ++s) {
            v16bf bk = *(const v16bf*)&Ks[nt * 16 + nl][s * 32 + 16 * half];
            acc = wmma_bf16(aq[s], bk, acc);
        }
#pragma unroll
        for (int r = 0; r < 8; ++r)
            Lg[8 * half + r][nt * 16 + nl] = acc[r] * 0.125f;  // 1/sqrt(64)
    }
    __syncthreads();

    // ---- softmax over each pixel's 7x7 window inside the halo ----
    if (t < 16) {
        int i = ti + (t >> 2), j = tj + (t & 3);
        int si = min(max(i - 3, 0), 57), sj = min(max(j - 3, 0), 57);
        int oy = si - hbi, ox = sj - hbj;       // in [0,3]
        float mx = -3.0e38f;
        for (int ky = 0; ky < 7; ++ky)
            for (int kx = 0; kx < 7; ++kx)
                mx = fmaxf(mx, Lg[t][(oy + ky) * 10 + ox + kx]);
        float sum = 0.0f;
        for (int ky = 0; ky < 7; ++ky)
            for (int kx = 0; kx < 7; ++kx)
                sum += __expf(Lg[t][(oy + ky) * 10 + ox + kx] - mx);
        float inv = 1.0f / sum;
        for (int ky = 0; ky < 7; ++ky)
            for (int kx = 0; kx < 7; ++kx) {
                int hp = (oy + ky) * 10 + ox + kx;
                At[t][hp] = (bf16)(__expf(Lg[t][hp] - mx) * inv);
            }
    }
    __syncthreads();

    // ---- out(16x64) = At(16x128) @ V(128x64); wave w -> dim tile w ----
    {
        const int nt = w;
        v8f acc = {0, 0, 0, 0, 0, 0, 0, 0};
#pragma unroll
        for (int ks = 0; ks < 4; ++ks) {
            v8bf lo = *(const v8bf*)&At[nl][ks * 32 + 16 * half];
            v8bf hi = *(const v8bf*)&At[nl][ks * 32 + 16 * half + 8];
            v16bf aa = combine16(lo, hi);
            v16bf bv = *(const v16bf*)&Vt[nt * 16 + nl][ks * 32 + 16 * half];
            acc = wmma_bf16(aa, bv, acc);
        }
#pragma unroll
        for (int r = 0; r < 8; ++r) {
            int m = 8 * half + r;
            int i = ti + (m >> 2), j = tj + (m & 3);
            long row = ((long)(b * 64 + i) * 64 + j);
            aout[row * 768 + cofs + nt * 16 + nl] = (bf16)acc[r];
        }
    }
}

// ---------------------------------------------------------------------------
// Launch
// ---------------------------------------------------------------------------
extern "C" void kernel_launch(void* const* d_in, const int* in_sizes, int n_in,
                              void* d_out, int out_size, void* d_ws, size_t ws_size,
                              hipStream_t stream) {
    (void)in_sizes; (void)n_in; (void)out_size; (void)ws_size;
    const float* x      = (const float*)d_in[0];   // [8,64,64,768]
    const float* w_qkv  = (const float*)d_in[1];   // [2304,768]
    const float* b_qkv  = (const float*)d_in[2];   // [2304]
    const float* w_proj = (const float*)d_in[3];   // [768,768]
    const float* b_proj = (const float*)d_in[4];   // [768]
    float* out = (float*)d_out;                    // [8,64,64,768]

    const long M = 8L * 64 * 64;                   // 32768 rows
    bf16* wqb  = (bf16*)d_ws;                      // 2304*768
    bf16* wpb  = wqb + 2304L * 768;                // 768*768
    bf16* qkvb = wpb + 768L * 768;                 // M*2304
    bf16* aob  = qkvb + M * 2304;                  // M*768  (total ~196 MB)

    // 1) weights fp32 -> bf16
    f32_to_bf16_kernel<<<(2304 * 768 + 255) / 256, 256, 0, stream>>>(
        w_qkv, wqb, 2304 * 768);
    f32_to_bf16_kernel<<<(768 * 768 + 255) / 256, 256, 0, stream>>>(
        w_proj, wpb, 768 * 768);

    // 2) QKV GEMM: [32768,768](f32, staged to bf16) x [2304,768]^T -> bf16
    gemm_wmma_kernel<true, false><<<dim3(2304 / 128, M / 128), 256, 0, stream>>>(
        (const void*)x, 768, wqb, 768, b_qkv, (void*)qkvb, 2304, 768);

    // 3) neighborhood attention: 8 * 12 * 16 * 16 blocks
    natten_wmma_kernel<<<8 * 12 * 16 * 16, 128, 0, stream>>>(qkvb, aob);

    // 4) output projection: [32768,768] x [768,768]^T + bias -> f32 out
    gemm_wmma_kernel<false, true><<<dim3(768 / 128, M / 128), 256, 0, stream>>>(
        (const void*)aob, 768, wpb, 768, b_proj, (void*)out, 768, 768);
}
